// GASTLCModel_69063074119817
// MI455X (gfx1250) — compile-verified, matrix operand order checked by reading, not run on hardware
//
#include <hip/hip_runtime.h>
#include <hip/hip_bf16.h>
#include <math.h>
#include <stdint.h>

// ---------------------------------------------------------------------------
// Types for WMMA / TDM
// ---------------------------------------------------------------------------
typedef __attribute__((ext_vector_type(8)))  float        f32x8;
typedef __attribute__((ext_vector_type(8)))  unsigned int u32x8;
typedef __attribute__((ext_vector_type(16))) __bf16       bf16x16;
typedef __attribute__((ext_vector_type(4)))  unsigned int u32x4;
typedef __attribute__((ext_vector_type(8)))  int          i32x8;
typedef __attribute__((ext_vector_type(4)))  int          i32x4;

#if defined(__AMDGCN__) && __has_builtin(__builtin_amdgcn_tensor_load_to_lds) && \
    __has_builtin(__builtin_amdgcn_s_wait_tensorcnt)
#define USE_TDM 1
#else
#define USE_TDM 0
#endif

__device__ __forceinline__ unsigned short f2bf(float f) {
    unsigned int u = __float_as_uint(f);
    u += 0x7fffu + ((u >> 16) & 1u);   // round-to-nearest-even
    return (unsigned short)(u >> 16);
}

__device__ __forceinline__ unsigned int pack_bf2(float a0, float a1) {
    return (unsigned int)f2bf(a0) | ((unsigned int)f2bf(a1) << 16);
}

#if USE_TDM
// Issue one TDM DMA: 2D f32 tile (tile1 rows x tile0 cols) from global to LDS.
// OOB reads (x >= tdim0 or y >= tdim1) return zero -> free K/N padding.
// LDS rows packed densely with stride tile0*4 bytes.
__device__ __forceinline__ void tdm_load_tile_f32(
    const float* gptr, unsigned lds_off, unsigned tdim0, unsigned tdim1,
    unsigned tile0, unsigned tile1, unsigned long long stride0_elems)
{
    unsigned long long ga = (unsigned long long)(uintptr_t)gptr;
    u32x4 g0;
    g0.x = 1u;                                            // count=1 (valid user D#)
    g0.y = lds_off;                                       // lds_addr [63:32]
    g0.z = (unsigned)(ga & 0xffffffffull);                // global_addr lo
    g0.w = (unsigned)((ga >> 32) & 0x1ffffffull) | (2u << 30);  // addr hi | type=2
    i32x8 g1;
    g1[0] = (int)(2u << 16);                              // data_size=2 (4 bytes)
    g1[1] = (int)((tdim0 & 0xffffu) << 16);               // tensor_dim0 [79:48]
    g1[2] = (int)((tdim0 >> 16) | ((tdim1 & 0xffffu) << 16));   // + tensor_dim1
    g1[3] = (int)((tdim1 >> 16) | (tile0 << 16));         // + tile_dim0 [127:112]
    g1[4] = (int)(tile1 & 0xffffu);                       // tile_dim1, tile_dim2=0
    g1[5] = (int)(unsigned)(stride0_elems & 0xffffffffull);      // stride0 lo
    g1[6] = (int)(unsigned)((stride0_elems >> 32) & 0xffffull);  // stride0 hi, stride1=0
    g1[7] = 0;
    i32x4 zz = {0, 0, 0, 0};
#if defined(__clang_major__) && (__clang_major__ >= 23)
    i32x8 z8 = {0, 0, 0, 0, 0, 0, 0, 0};
    __builtin_amdgcn_tensor_load_to_lds(g0, g1, zz, zz, z8, 0);
#else
    __builtin_amdgcn_tensor_load_to_lds(g0, g1, zz, zz, 0);
#endif
}
#endif

// ---------------------------------------------------------------------------
// Utility kernels
// ---------------------------------------------------------------------------
__global__ void zero32_kernel(unsigned int* p, long long n) {
    long long i = (long long)blockIdx.x * blockDim.x + threadIdx.x;
    if (i < n) p[i] = 0u;
}

__global__ void deg_kernel(const int* __restrict__ src, const int* __restrict__ dst,
                           float* __restrict__ deg, int E) {
    int e = blockIdx.x * blockDim.x + threadIdx.x;
    if (e < E) {
        if (src[e] != dst[e]) atomicAdd(&deg[dst[e]], 1.0f);
    }
}

__global__ void node_norm_kernel(const float* __restrict__ deg,
                                 float* __restrict__ dis, float* __restrict__ selfn, int N) {
    int n = blockIdx.x * blockDim.x + threadIdx.x;
    if (n < N) {
        float d  = deg[n] + 1.0f;
        float rs = rsqrtf(d);
        dis[n]   = rs;
        selfn[n] = rs * rs;
    }
}

__global__ void edge_norm_kernel(const int* __restrict__ src, const int* __restrict__ dst,
                                 const float* __restrict__ dis, float* __restrict__ norm, int E) {
    int e = blockIdx.x * blockDim.x + threadIdx.x;
    if (e < E) {
        int s = src[e], d = dst[e];
        norm[e] = (s != d) ? dis[s] * dis[d] : 0.0f;
    }
}

// ---------------------------------------------------------------------------
// H = X(NxK) @ W(Kx128) via v_wmma_f32_16x16x32_bf16.
// One block = 256 threads (8 waves) computes a 128x128 output tile.
//  - X tile staged as f32 by one Tensor-Data-Mover DMA (zero-padded to
//    128x128 via the D# tensor dims), then converted once LDS->LDS to bf16.
//  - W staged once as bf16 (transposed) in LDS.
//  - Inner loop is pure ds_load_b128 + v_wmma (fully unrolled, 32 wmma).
// ---------------------------------------------------------------------------
__global__ __launch_bounds__(256) void gemm_wmma_kernel(
    const float* __restrict__ X, int ldx, int K,
    const float* __restrict__ W, float* __restrict__ H, int N)
{
    __shared__ unsigned short sW[128 * 128];   // bf16 [n][k] (transposed), 32KB
    __shared__ float          sXf[128 * 128];  // f32  [m][k], 64KB (TDM target)
    __shared__ unsigned short sXb[128 * 128];  // bf16 [m][k], 32KB

    const int tid   = threadIdx.x;
    const int wave  = tid >> 5;
    const int lane  = tid & 31;
    const int mLane = lane & 15;
    const int khalf = lane >> 4;
    const int rowBlock = blockIdx.x * 128;

    // Speculative prefetch of the next block's X tile (global_prefetch_b8)
    if (tid == 0)
        __builtin_prefetch(X + (size_t)(rowBlock + 128) * ldx, 0, 1);

    // Stage W transposed into LDS as bf16 (zero-pad k >= K)
    for (int i = tid; i < 128 * 128; i += 256) {
        int n = i >> 7, k = i & 127;
        float w = (k < K) ? W[k * 128 + n] : 0.0f;
        sW[n * 128 + k] = f2bf(w);
    }

#if USE_TDM
    // One TDM DMA for the whole 128x128 X tile; zero-filled beyond K / N.
    if (wave == 0) {
        tdm_load_tile_f32(X + (size_t)rowBlock * ldx,
                          (unsigned)(uintptr_t)sXf,
                          (unsigned)K, (unsigned)(N - rowBlock),
                          128u, 128u, (unsigned long long)ldx);
        __builtin_amdgcn_s_wait_tensorcnt(0);
    }
#else
    for (int i = tid; i < 128 * 128; i += 256) {
        int m = i >> 7, k = i & 127;
        int row = rowBlock + m;
        sXf[i] = (row < N && k < K) ? X[(size_t)row * ldx + k] : 0.0f;
    }
#endif
    __syncthreads();

    // One-shot LDS->LDS f32 -> packed bf16 conversion (out of the hot loop)
    {
        unsigned int* dst32 = (unsigned int*)sXb;
        for (int i = tid; i < 128 * 64; i += 256) {
            int e = i * 2;
            dst32[i] = pack_bf2(sXf[e], sXf[e + 1]);
        }
    }
    __syncthreads();

    f32x8 acc[8] = {};
    const int mWave = wave * 16;
    const unsigned short* arow = &sXb[(mWave + mLane) * 128];

#pragma unroll
    for (int kc = 0; kc < 128; kc += 32) {
        // A fragment (16x32 bf16), ISA 7.12.2 layout: two b128 LDS loads
        u32x8 araw;
#pragma unroll
        for (int v = 0; v < 8; ++v) {
            int k = kc + ((v & 3) * 2) + ((v >= 4) ? 16 : 0) + khalf * 8;
            araw[v] = *(const unsigned int*)&arow[k];
        }
        bf16x16 afrag = __builtin_bit_cast(bf16x16, araw);

#pragma unroll
        for (int ct = 0; ct < 8; ++ct) {
            u32x8 braw;
#pragma unroll
            for (int v = 0; v < 8; ++v) {
                int k = ((v & 3) * 2) + ((v >= 4) ? 16 : 0) + khalf * 8;
                braw[v] = *(const unsigned int*)&sW[(ct * 16 + mLane) * 128 + (kc + k)];
            }
            bf16x16 bfrag = __builtin_bit_cast(bf16x16, braw);
            acc[ct] = __builtin_amdgcn_wmma_f32_16x16x32_bf16(
                false, afrag, false, bfrag, (short)0, acc[ct], false, false);
        }
    }

    // Store D: VGPR r -> (M = r + 8*khalf, N = lane%16)
    if (rowBlock + 128 <= N) {
        // full tile: unconditional coalesced stores
#pragma unroll
        for (int ct = 0; ct < 8; ++ct) {
            float* hp = &H[(size_t)(rowBlock + mWave + khalf * 8) * 128 + ct * 16 + mLane];
#pragma unroll
            for (int r = 0; r < 8; ++r) hp[(size_t)r * 128] = acc[ct][r];
        }
    } else {
#pragma unroll
        for (int ct = 0; ct < 8; ++ct) {
#pragma unroll
            for (int r = 0; r < 8; ++r) {
                int row = rowBlock + mWave + r + khalf * 8;
                int col = ct * 16 + mLane;
                if (row < N) H[(size_t)row * 128 + col] = acc[ct][r];
            }
        }
    }
}

// acc[n,f] = h[n,f]*self_norm[n] + b[f]
__global__ void init_acc_kernel(const float* __restrict__ h, const float* __restrict__ selfn,
                                const float* __restrict__ b, float* __restrict__ acc, int N) {
    long long i = (long long)blockIdx.x * blockDim.x + threadIdx.x;
    if (i < (long long)N * 128) {
        int n = (int)(i >> 7), f = (int)(i & 127);
        acc[i] = h[i] * selfn[n] + b[f];
    }
}

// acc[dst] += h[src] * norm  (one wave per edge, float4 per lane)
__global__ void scatter_feat_kernel(const float* __restrict__ h, const float* __restrict__ norm,
                                    const int* __restrict__ src, const int* __restrict__ dst,
                                    float* __restrict__ acc, int E) {
    long long idx = (long long)blockIdx.x * blockDim.x + threadIdx.x;
    int e = (int)(idx >> 5);
    int lane = (int)(idx & 31);
    if (e >= E) return;
    float w = norm[e];
    if (w == 0.0f) return;
    const float4 hv = *(const float4*)&h[(size_t)src[e] * 128 + lane * 4];
    float* a = &acc[(size_t)dst[e] * 128 + lane * 4];
    atomicAdd(a + 0, hv.x * w);
    atomicAdd(a + 1, hv.y * w);
    atomicAdd(a + 2, hv.z * w);
    atomicAdd(a + 3, hv.w * w);
}

__global__ void tanh_kernel(float* x, long long n) {
    long long i = (long long)blockIdx.x * blockDim.x + threadIdx.x;
    if (i < n) x[i] = tanhf(x[i]);
}

// Layer 4: h4[n] = x3[n,:] . W4 ;  x4acc[n] = h4*self_norm + b4
__global__ void dot_w4_kernel(const float* __restrict__ x3, const float* __restrict__ W4,
                              const float* __restrict__ b4, const float* __restrict__ selfn,
                              float* __restrict__ h4, float* __restrict__ x4acc, int N) {
    int n = blockIdx.x * (blockDim.x >> 5) + (threadIdx.x >> 5);
    int lane = threadIdx.x & 31;
    if (n >= N) return;
    float s = 0.0f;
    for (int k = lane; k < 128; k += 32) s += x3[(size_t)n * 128 + k] * W4[k];
    for (int off = 16; off; off >>= 1) s += __shfl_down(s, off);
    if (lane == 0) {
        h4[n] = s;
        x4acc[n] = s * selfn[n] + b4[0];
    }
}

__global__ void scatter_scalar_kernel(const float* __restrict__ h4, const float* __restrict__ norm,
                                      const int* __restrict__ src, const int* __restrict__ dst,
                                      float* __restrict__ acc, int E) {
    int e = blockIdx.x * blockDim.x + threadIdx.x;
    if (e < E) {
        float w = norm[e];
        if (w != 0.0f) atomicAdd(&acc[dst[e]], h4[src[e]] * w);
    }
}

__global__ void count_kernel(const int* __restrict__ batch, int* __restrict__ counts, int N) {
    int i = blockIdx.x * blockDim.x + threadIdx.x;
    if (i < N) atomicAdd(&counts[batch[i]], 1);
}

__global__ void starts_kernel(const int* __restrict__ counts, int* __restrict__ starts) {
    if (threadIdx.x == 0 && blockIdx.x == 0) {
        int s = 0;
        for (int g = 0; g < 11; ++g) { starts[g] = s; s += counts[g]; }
    }
}

// ---------------------------------------------------------------------------
// SortPool: per graph select top-64 by (score desc, index asc), copy
// concat(x1,x2,x3,x4) rows (385) into pooled[g][rank][:]. One block/graph.
// ---------------------------------------------------------------------------
__global__ __launch_bounds__(256) void topk_pool_kernel(
    const float* __restrict__ x1, const float* __restrict__ x2,
    const float* __restrict__ x3, const float* __restrict__ x4,
    const int* __restrict__ starts, const int* __restrict__ counts,
    float* __restrict__ pooled)
{
    const int g = blockIdx.x;
    const int start = starts[g];
    const int cnt   = counts[g];
    __shared__ float sS[256];
    __shared__ int   sI[256];
    __shared__ float thS;
    __shared__ int   thI;
    __shared__ int   selNode;
    if (threadIdx.x == 0) { thS = INFINITY; thI = -1; }
    __syncthreads();

    for (int r = 0; r < 64; ++r) {
        float tS = thS; int tI = thI;
        float bestS = -INFINITY; int bestI = 0x7fffffff;
        for (int i = threadIdx.x; i < cnt; i += 256) {
            int n = start + i;
            float s = x4[n];
            // strictly after threshold in (score desc, index asc) order
            if (s < tS || (s == tS && n > tI)) {
                if (s > bestS || (s == bestS && n < bestI)) { bestS = s; bestI = n; }
            }
        }
        sS[threadIdx.x] = bestS; sI[threadIdx.x] = bestI;
        __syncthreads();
        for (int off = 128; off; off >>= 1) {
            if (threadIdx.x < off) {
                float s2 = sS[threadIdx.x + off]; int i2 = sI[threadIdx.x + off];
                if (s2 > sS[threadIdx.x] || (s2 == sS[threadIdx.x] && i2 < sI[threadIdx.x])) {
                    sS[threadIdx.x] = s2; sI[threadIdx.x] = i2;
                }
            }
            __syncthreads();
        }
        if (threadIdx.x == 0) {
            selNode = (sI[0] != 0x7fffffff) ? sI[0] : -1;
            thS = sS[0]; thI = sI[0];
        }
        __syncthreads();
        int node = selNode;
        if (node >= 0) {
            float* row = &pooled[((size_t)g * 64 + r) * 385];
            for (int k = threadIdx.x; k < 385; k += 256) {
                float v;
                if      (k < 128) v = x1[(size_t)node * 128 + k];
                else if (k < 256) v = x2[(size_t)node * 128 + (k - 128)];
                else if (k < 384) v = x3[(size_t)node * 128 + (k - 256)];
                else              v = x4[node];
                row[k] = v;
            }
        }
        __syncthreads();
    }
}

// conv5: out[g][c][t] = relu(cb5[c] + sum_k pooled[g][t][k]*cw5[c][k]), stride=385
__global__ void conv5_kernel(const float* __restrict__ pooled, const float* __restrict__ cw5,
                             const float* __restrict__ cb5, float* __restrict__ out) {
    int g = blockIdx.x, c = blockIdx.y, t = threadIdx.x;  // t < 64
    const float* in = &pooled[((size_t)g * 64 + t) * 385];
    const float* w  = &cw5[(size_t)c * 385];
    float s = cb5[c];
    for (int k = 0; k < 385; ++k) s += in[k] * w[k];
    out[((size_t)g * 64 + c) * 64 + t] = fmaxf(s, 0.0f);
}

__global__ void maxpool_kernel(const float* __restrict__ in, float* __restrict__ out) {
    int i = blockIdx.x * blockDim.x + threadIdx.x;  // G*64*32
    if (i < 11 * 64 * 32) {
        int j = i & 31, gc = i >> 5;
        out[i] = fmaxf(in[gc * 64 + 2 * j], in[gc * 64 + 2 * j + 1]);
    }
}

// conv6: (G,64,32) -> (G,128,28), kernel 5
__global__ void conv6_kernel(const float* __restrict__ in, const float* __restrict__ cw6,
                             const float* __restrict__ cb6, float* __restrict__ out) {
    int g = blockIdx.x, co = blockIdx.y, t = threadIdx.x;
    if (t >= 28) return;
    float s = cb6[co];
    for (int ci = 0; ci < 64; ++ci) {
        const float* ip = &in[((size_t)g * 64 + ci) * 32 + t];
        const float* wp = &cw6[((size_t)co * 64 + ci) * 5];
#pragma unroll
        for (int k = 0; k < 5; ++k) s += ip[k] * wp[k];
    }
    out[((size_t)g * 128 + co) * 28 + t] = fmaxf(s, 0.0f);
}

__global__ __launch_bounds__(256) void dense1_kernel(const float* __restrict__ emb,
                                                     const float* __restrict__ fw1,
                                                     const float* __restrict__ fb1,
                                                     float* __restrict__ out) {
    const int j = blockIdx.x;  // 128 outputs
    __shared__ float red[256];
    const float* w = &fw1[(size_t)j * (11 * 3584)];
    float s = 0.0f;
    for (int i = threadIdx.x; i < 11 * 3584; i += 256) s += emb[i] * w[i];
    red[threadIdx.x] = s;
    __syncthreads();
    for (int off = 128; off; off >>= 1) {
        if (threadIdx.x < off) red[threadIdx.x] += red[threadIdx.x + off];
        __syncthreads();
    }
    if (threadIdx.x == 0) out[j] = fmaxf(red[0] + fb1[j], 0.0f);
}

__global__ void dense2_kernel(const float* __restrict__ in, const float* __restrict__ fw2,
                              const float* __restrict__ fb2, float* __restrict__ out) {
    int j = threadIdx.x;
    if (j < 10) {
        float s = fb2[j];
        for (int i = 0; i < 128; ++i) s += in[i] * fw2[j * 128 + i];
        out[j] = s;
    }
}

// ---------------------------------------------------------------------------
// Host driver
// ---------------------------------------------------------------------------
extern "C" void kernel_launch(void* const* d_in, const int* in_sizes, int n_in,
                              void* d_out, int out_size, void* d_ws, size_t ws_size,
                              hipStream_t stream) {
    const float* x     = (const float*)d_in[0];
    const int*   ei    = (const int*)d_in[1];
    const int*   batch = (const int*)d_in[2];
    const float* W1 = (const float*)d_in[3];  const float* b1 = (const float*)d_in[4];
    const float* W2 = (const float*)d_in[5];  const float* b2 = (const float*)d_in[6];
    const float* W3 = (const float*)d_in[7];  const float* b3 = (const float*)d_in[8];
    const float* W4 = (const float*)d_in[9];  const float* b4 = (const float*)d_in[10];
    const float* cw5 = (const float*)d_in[11]; const float* cb5 = (const float*)d_in[12];
    const float* cw6 = (const float*)d_in[13]; const float* cb6 = (const float*)d_in[14];
    const float* fw1 = (const float*)d_in[15]; const float* fb1 = (const float*)d_in[16];
    const float* fw2 = (const float*)d_in[17]; const float* fb2 = (const float*)d_in[18];
    float* out = (float*)d_out;

    const int N = in_sizes[2];         // 400000
    const int E = in_sizes[1] / 2;     // 1600000
    const int* src = ei;
    const int* dst = ei + E;

    // ---- workspace carving ----
    char* base = (char*)d_ws;
    size_t off = 0;
    auto alloc = [&](size_t bytes) -> void* {
        void* p = base + off;
        off = (off + bytes + 255) & ~(size_t)255;
        return p;
    };
    float* deg   = (float*)alloc((size_t)N * 4);
    float* dis   = (float*)alloc((size_t)N * 4);
    float* selfn = (float*)alloc((size_t)N * 4);
    float* norm  = (float*)alloc((size_t)E * 4);
    float* h     = (float*)alloc((size_t)N * 128 * 4);
    float* x1    = (float*)alloc((size_t)N * 128 * 4);
    float* x2    = (float*)alloc((size_t)N * 128 * 4);
    float* x3    = (float*)alloc((size_t)N * 128 * 4);
    float* x4    = (float*)alloc((size_t)N * 4);
    float* h4    = (float*)alloc((size_t)N * 4);
    int*   counts = (int*)alloc(11 * 4);
    int*   starts = (int*)alloc(11 * 4);
    float* pooled = (float*)alloc((size_t)11 * 64 * 385 * 4);
    float* c5out  = (float*)alloc((size_t)11 * 64 * 64 * 4);
    float* pool2  = (float*)alloc((size_t)11 * 64 * 32 * 4);
    float* c6out  = (float*)alloc((size_t)11 * 128 * 28 * 4);
    float* d1out  = (float*)alloc(128 * 4);

    const int T = 256;
    auto blocks = [&](long long n) { return (unsigned)((n + T - 1) / T); };

    // ---- normalization ----
    zero32_kernel<<<blocks(N), T, 0, stream>>>((unsigned int*)deg, N);
    deg_kernel<<<blocks(E), T, 0, stream>>>(src, dst, deg, E);
    node_norm_kernel<<<blocks(N), T, 0, stream>>>(deg, dis, selfn, N);
    edge_norm_kernel<<<blocks(E), T, 0, stream>>>(src, dst, dis, norm, E);

    const unsigned gemmBlocks = (unsigned)((N + 127) / 128);
    const long long NF = (long long)N * 128;
    const long long EL = (long long)E * 32;

    // ---- GCN layer 1 (K=100) ----
    gemm_wmma_kernel<<<gemmBlocks, 256, 0, stream>>>(x, 100, 100, W1, h, N);
    init_acc_kernel<<<blocks(NF), T, 0, stream>>>(h, selfn, b1, x1, N);
    scatter_feat_kernel<<<blocks(EL), T, 0, stream>>>(h, norm, src, dst, x1, E);
    tanh_kernel<<<blocks(NF), T, 0, stream>>>(x1, NF);

    // ---- GCN layer 2 ----
    gemm_wmma_kernel<<<gemmBlocks, 256, 0, stream>>>(x1, 128, 128, W2, h, N);
    init_acc_kernel<<<blocks(NF), T, 0, stream>>>(h, selfn, b2, x2, N);
    scatter_feat_kernel<<<blocks(EL), T, 0, stream>>>(h, norm, src, dst, x2, E);
    tanh_kernel<<<blocks(NF), T, 0, stream>>>(x2, NF);

    // ---- GCN layer 3 ----
    gemm_wmma_kernel<<<gemmBlocks, 256, 0, stream>>>(x2, 128, 128, W3, h, N);
    init_acc_kernel<<<blocks(NF), T, 0, stream>>>(h, selfn, b3, x3, N);
    scatter_feat_kernel<<<blocks(EL), T, 0, stream>>>(h, norm, src, dst, x3, E);
    tanh_kernel<<<blocks(NF), T, 0, stream>>>(x3, NF);

    // ---- GCN layer 4 (128 -> 1) ----
    dot_w4_kernel<<<blocks((long long)N * 32), T, 0, stream>>>(x3, W4, b4, selfn, h4, x4, N);
    scatter_scalar_kernel<<<blocks(E), T, 0, stream>>>(h4, norm, src, dst, x4, E);
    tanh_kernel<<<blocks(N), T, 0, stream>>>(x4, N);

    // ---- SortPool ----
    zero32_kernel<<<1, 64, 0, stream>>>((unsigned int*)counts, 11);
    count_kernel<<<blocks(N), T, 0, stream>>>(batch, counts, N);
    starts_kernel<<<1, 1, 0, stream>>>(counts, starts);
    zero32_kernel<<<blocks((long long)11 * 64 * 385), T, 0, stream>>>((unsigned int*)pooled,
                                                                      (long long)11 * 64 * 385);
    topk_pool_kernel<<<11, 256, 0, stream>>>(x1, x2, x3, x4, starts, counts, pooled);

    // ---- conv / pool / dense head ----
    conv5_kernel<<<dim3(11, 64), 64, 0, stream>>>(pooled, cw5, cb5, c5out);
    maxpool_kernel<<<blocks(11 * 64 * 32), T, 0, stream>>>(c5out, pool2);
    conv6_kernel<<<dim3(11, 128), 32, 0, stream>>>(pool2, cw6, cb6, c6out);
    dense1_kernel<<<128, 256, 0, stream>>>(c6out, fw1, fb1, d1out);
    dense2_kernel<<<1, 32, 0, stream>>>(d1out, fw2, fb2, out);
}
